// PolarizedMamba2Block_82927228551509
// MI455X (gfx1250) — compile-verified, matrix operand order checked by reading, not run on hardware
//
#include <hip/hip_runtime.h>
#include <hip/hip_bf16.h>

// ---------------------------------------------------------------------------
// PolarizedMamba2Block for MI455X (gfx1250, wave32, WMMA).
// GEMMs (~122 GFLOP) run on v_wmma_f32_16x16x32_bf16 with f32 accumulation.
// 32x32 wave tile (4 accumulators) + double-buffered K pipeline so WMMAs on
// slab k overlap the global loads of slab k+32 (avoids s_wait_loadcnt 0
// directly in front of every WMMA). Weights (~30MB bf16) live in the 192MB
// L2. SSM scan stages B_t/C_t via global_load_async_to_lds_b32 (ASYNCcnt).
// ---------------------------------------------------------------------------

#define DEVI __device__ __forceinline__

typedef __attribute__((ext_vector_type(8)))  float  v8f;
typedef __attribute__((ext_vector_type(8)))  __bf16 v8bf;
typedef __attribute__((ext_vector_type(16))) __bf16 v16bf;

constexpr int D_MODEL   = 1024;
constexpr int D_STATE   = 128;
constexpr int D_CONV    = 4;
constexpr int HEADDIM   = 64;
constexpr int D_INNER   = 2048;
constexpr int NHEADS    = 32;
constexpr int D_IN_PROJ = 2 * D_INNER + 2 * D_STATE + NHEADS;  // 4384
constexpr int CONV_CH   = D_INNER + 2 * D_STATE;               // 2304
constexpr int BATCH     = 2;
constexpr int SEQ       = 2048;
constexpr int ROWS      = BATCH * SEQ;                         // 4096
constexpr int DFUSE     = 3 * D_INNER;                         // 6144

DEVI unsigned short f32_to_bf16(float f) {
    union { float f; unsigned u; } v{f};
    unsigned r = v.u + 0x7fffu + ((v.u >> 16) & 1u);   // round-to-nearest-even
    return (unsigned short)(r >> 16);
}
DEVI float silu_f(float x) { return x / (1.0f + __expf(-x)); }

DEVI v16bf load_frag(const unsigned short* p) {
    v8bf lo = *(const v8bf*)(p);
    v8bf hi = *(const v8bf*)(p + 16);
    v16bf r;
#pragma unroll
    for (int i = 0; i < 8; ++i) { r[i] = lo[i]; r[i + 8] = hi[i]; }
    return r;
}

#define WMMA_BF16(A_, B_, C_) \
    __builtin_amdgcn_wmma_f32_16x16x32_bf16(false, (A_), false, (B_), (short)0, (C_), false, false)

// ---------------------------------------------------------------------------
// bf16 WMMA GEMM:  C[M,N] = A[M,K] * W[N,K]^T   (both K-contiguous, bf16)
// Wave tile: 32(M) x 32(N): 2 A-frags x 2 B-frags -> 4 WMMA per K-slab of 32.
// Double-buffered over K (K % 64 == 0 for all calls) so loads overlap WMMA.
// Block: 8 waves stacked in M -> block tile 256(M) x 32(N).
// grid = (N/32, M/256). All sizes exact multiples -> EXEC all-ones for WMMA.
// ---------------------------------------------------------------------------
__global__ void __launch_bounds__(256) gemm_bf16_wmma(
    const unsigned short* __restrict__ A,
    const unsigned short* __restrict__ W,
    int K,
    float* __restrict__ Cf,                 // f32 out (or nullptr)
    unsigned short* __restrict__ Cb,        // bf16 out (used if Cf==nullptr)
    int ldc, int coff,
    const float* __restrict__ resid)        // optional residual, ld == ldc
{
    const int tid  = threadIdx.x;
    const int wave = tid >> 5;
    const int lane = tid & 31;
    const int half = lane >> 4;   // K-half selector (CDNA5 16-bit A/B layout)
    const int l16  = lane & 15;
    const int m0 = blockIdx.y * 256 + wave * 32;
    const int n0 = blockIdx.x * 32;

    const unsigned short* pa0 = A + (size_t)(m0 + l16) * K + half * 8;
    const unsigned short* pa1 = pa0 + (size_t)16 * K;
    const unsigned short* pb0 = W + (size_t)(n0 + l16) * K + half * 8;
    const unsigned short* pb1 = pb0 + (size_t)16 * K;

    v8f c00 = {}, c01 = {}, c10 = {}, c11 = {};

    // stage 0 preload
    v16bf a0 = load_frag(pa0), a1 = load_frag(pa1);
    v16bf b0 = load_frag(pb0), b1 = load_frag(pb1);

    for (int k0 = 0; k0 < K; k0 += 64) {
        if (k0 + 128 < K) {                  // global_prefetch_b8 two slabs out
            __builtin_prefetch(pa0 + 128, 0, 1);
            __builtin_prefetch(pa1 + 128, 0, 1);
            __builtin_prefetch(pb0 + 128, 0, 1);
            __builtin_prefetch(pb1 + 128, 0, 1);
        }
        // issue loads for slab k0+32 while computing slab k0
        v16bf a0b = load_frag(pa0 + 32), a1b = load_frag(pa1 + 32);
        v16bf b0b = load_frag(pb0 + 32), b1b = load_frag(pb1 + 32);
        c00 = WMMA_BF16(a0, b0, c00);
        c01 = WMMA_BF16(a0, b1, c01);
        c10 = WMMA_BF16(a1, b0, c10);
        c11 = WMMA_BF16(a1, b1, c11);
        pa0 += 64; pa1 += 64; pb0 += 64; pb1 += 64;
        if (k0 + 64 < K) {                   // loads for slab k0+64 overlap b-WMMAs
            a0 = load_frag(pa0); a1 = load_frag(pa1);
            b0 = load_frag(pb0); b1 = load_frag(pb1);
        }
        c00 = WMMA_BF16(a0b, b0b, c00);
        c01 = WMMA_BF16(a0b, b1b, c01);
        c10 = WMMA_BF16(a1b, b0b, c10);
        c11 = WMMA_BF16(a1b, b1b, c11);
    }

#pragma unroll
    for (int r = 0; r < 8; ++r) {
        const int row0 = m0 + r + half * 8;          // C/D layout: M = r + 8*half
        const int row1 = row0 + 16;
        const int col0 = n0 + l16;
        const int col1 = col0 + 16;
        if (Cf) {
            float v00 = c00[r], v01 = c01[r], v10 = c10[r], v11 = c11[r];
            if (resid) {
                v00 += resid[(size_t)row0 * ldc + col0];
                v01 += resid[(size_t)row0 * ldc + col1];
                v10 += resid[(size_t)row1 * ldc + col0];
                v11 += resid[(size_t)row1 * ldc + col1];
            }
            Cf[(size_t)row0 * ldc + coff + col0] = v00;
            Cf[(size_t)row0 * ldc + coff + col1] = v01;
            Cf[(size_t)row1 * ldc + coff + col0] = v10;
            Cf[(size_t)row1 * ldc + coff + col1] = v11;
        } else {
            Cb[(size_t)row0 * ldc + coff + col0] = f32_to_bf16(c00[r]);
            Cb[(size_t)row0 * ldc + coff + col1] = f32_to_bf16(c01[r]);
            Cb[(size_t)row1 * ldc + coff + col0] = f32_to_bf16(c10[r]);
            Cb[(size_t)row1 * ldc + coff + col1] = f32_to_bf16(c11[r]);
        }
    }
}

// ---------------------------------------------------------------------------
// Streaming / pointwise kernels
// ---------------------------------------------------------------------------
__global__ void cvt_bf16_kernel(const float* __restrict__ in,
                                unsigned short* __restrict__ out, int n)
{
    int i = blockIdx.x * blockDim.x + threadIdx.x;
    if (i < n) out[i] = f32_to_bf16(in[i]);
}

// rmsnorm(x)*w over D_MODEL; emits f32 (for conv/scan path math) + bf16 (GEMM A)
__global__ void __launch_bounds__(256) rmsnorm_in_kernel(
    const float* __restrict__ x, const float* __restrict__ w,
    float* __restrict__ xn, unsigned short* __restrict__ xnb)
{
    __shared__ float red[256];
    const int row = blockIdx.x, tid = threadIdx.x;
    const float* xr = x + (size_t)row * D_MODEL;
    float v[4]; float s = 0.f;
#pragma unroll
    for (int i = 0; i < 4; ++i) { v[i] = xr[tid + i * 256]; s += v[i] * v[i]; }
    red[tid] = s; __syncthreads();
    for (int st = 128; st > 0; st >>= 1) {
        if (tid < st) red[tid] += red[tid + st];
        __syncthreads();
    }
    const float scale = rsqrtf(red[0] / D_MODEL + 1e-5f);
#pragma unroll
    for (int i = 0; i < 4; ++i) {
        int j = tid + i * 256;
        float o = v[i] * scale * w[j];
        xn [(size_t)row * D_MODEL + j] = o;
        xnb[(size_t)row * D_MODEL + j] = f32_to_bf16(o);
    }
}

// depthwise causal conv (k=4) + bias + SiLU over zxbcdt cols [D_INNER, D_INNER+CONV_CH)
__global__ void conv_silu_kernel(const float* __restrict__ zx,
                                 const float* __restrict__ cw,
                                 const float* __restrict__ cb,
                                 float* __restrict__ xbc)
{
    size_t idx = (size_t)blockIdx.x * blockDim.x + threadIdx.x;
    if (idx >= (size_t)ROWS * CONV_CH) return;
    const int c  = (int)(idx % CONV_CH);
    const int rt = (int)(idx / CONV_CH);
    const int t  = rt % SEQ;
    const int b  = rt / SEQ;
    float acc = cb[c];
#pragma unroll
    for (int j = 0; j < D_CONV; ++j) {
        int tt = t - (D_CONV - 1) + j;
        if (tt >= 0)
            acc += cw[c * D_CONV + j] *
                   zx[(size_t)(b * SEQ + tt) * D_IN_PROJ + D_INNER + c];
    }
    xbc[idx] = silu_f(acc);
}

// dt = softplus(dt_raw + bias); dA = exp(-exp(A_log)*dt)
__global__ void dtda_kernel(const float* __restrict__ zx,
                            const float* __restrict__ dt_bias,
                            const float* __restrict__ A_log,
                            float* __restrict__ dt, float* __restrict__ dA)
{
    int idx = blockIdx.x * blockDim.x + threadIdx.x;
    if (idx >= ROWS * NHEADS) return;
    const int h   = idx & (NHEADS - 1);
    const int row = idx >> 5;
    float v  = zx[(size_t)row * D_IN_PROJ + D_INNER + CONV_CH + h] + dt_bias[h];
    float sp = (v > 20.f) ? v : log1pf(__expf(v));
    dt[idx] = sp;
    dA[idx] = __expf(-__expf(A_log[h]) * sp);
}

// Sequential SSM scan: one workgroup per (batch, head).
// State h[64][128] f32 held in registers (32 per thread).
// B_t/C_t staged memory->LDS via global_load_async_to_lds_b32 (ASYNCcnt).
__global__ void __launch_bounds__(256) ssm_scan_kernel(
    const float* __restrict__ xbc, const float* __restrict__ dt,
    const float* __restrict__ dA, const float* __restrict__ Dw,
    float* __restrict__ y)
{
    __shared__ float BCs[2 * D_STATE];   // [0:128)=B_t, [128:256)=C_t
    __shared__ float red[256];
    const int b   = blockIdx.x >> 5;
    const int h   = blockIdx.x & 31;
    const int tid = threadIdx.x;
    const int p   = tid >> 2;        // head-dim index 0..63
    const int g   = tid & 3;         // state-chunk 0..3 (32 n-values each)
    const int nb  = g * 32;
    const float Dh = Dw[h];
    const unsigned lds_off = (unsigned)(size_t)(&BCs[tid]);  // LDS byte addr (low 32b)
    float hreg[32];
#pragma unroll
    for (int i = 0; i < 32; ++i) hreg[i] = 0.f;

    for (int t = 0; t < SEQ; ++t) {
        const int row = b * SEQ + t;
        const size_t rb = (size_t)row * CONV_CH;
        {   // async stage of B_t (128) + C_t (128): one b32 per lane, no VGPR bounce
            const float* gp = xbc + rb + D_INNER + tid;
            asm volatile("global_load_async_to_lds_b32 %0, %1, off"
                         :: "v"(lds_off), "v"(gp) : "memory");
            asm volatile("s_wait_asynccnt 0x0" ::: "memory");
        }
        __syncthreads();
        const float dA_t = dA[row * NHEADS + h];
        const float dt_t = dt[row * NHEADS + h];
        const float x_t  = xbc[rb + h * HEADDIM + p];
        const float dtx  = dt_t * x_t;
        float acc = 0.f;
#pragma unroll
        for (int i = 0; i < 32; ++i) {
            float hv = hreg[i] * dA_t + dtx * BCs[nb + i];
            hreg[i] = hv;
            acc += hv * BCs[D_STATE + nb + i];
        }
        red[tid] = acc; __syncthreads();
        if (g == 0) {
            float ys = red[tid] + red[tid + 1] + red[tid + 2] + red[tid + 3]
                     + Dh * x_t;
            y[(size_t)row * D_INNER + h * HEADDIM + p] = ys;
        }
        __syncthreads();
    }
}

// fused_out[:, 0:2048) = bf16( rmsnorm( y * silu(z), gate_norm_w ) )
__global__ void __launch_bounds__(256) gate_norm_kernel(
    const float* __restrict__ y, const float* __restrict__ zx,
    const float* __restrict__ gw, unsigned short* __restrict__ fused)
{
    __shared__ float vals[D_INNER];
    __shared__ float red[256];
    const int row = blockIdx.x, tid = threadIdx.x;
    float s = 0.f;
#pragma unroll
    for (int i = 0; i < 8; ++i) {
        int j = tid + i * 256;
        float z = zx[(size_t)row * D_IN_PROJ + j];
        float v = y[(size_t)row * D_INNER + j] * silu_f(z);
        vals[j] = v; s += v * v;
    }
    red[tid] = s; __syncthreads();
    for (int st = 128; st > 0; st >>= 1) {
        if (tid < st) red[tid] += red[tid + st];
        __syncthreads();
    }
    const float scale = rsqrtf(red[0] / D_INNER + 1e-5f);
#pragma unroll
    for (int i = 0; i < 8; ++i) {
        int j = tid + i * 256;
        fused[(size_t)row * DFUSE + j] = f32_to_bf16(vals[j] * scale * gw[j]);
    }
}

// cumsum over T per (batch, channel); coalesced across channels at each t
__global__ void cumsum_kernel(const float* __restrict__ src,
                              unsigned short* __restrict__ fused)
{
    int idx = blockIdx.x * blockDim.x + threadIdx.x;
    if (idx >= BATCH * D_INNER) return;
    const int b = idx / D_INNER, c = idx % D_INNER;
    float acc = 0.f;
    for (int t = 0; t < SEQ; ++t) {
        size_t row = (size_t)(b * SEQ + t);
        acc += src[row * D_INNER + c];
        fused[row * DFUSE + 2 * D_INNER + c] = f32_to_bf16(acc);
    }
}

// ---------------------------------------------------------------------------
// Host side
// ---------------------------------------------------------------------------
static constexpr size_t al256(size_t x) { return (x + 255) & ~size_t(255); }

extern "C" void kernel_launch(void* const* d_in, const int* in_sizes, int n_in,
                              void* d_out, int out_size, void* d_ws, size_t ws_size,
                              hipStream_t stream) {
    const float* x        = (const float*)d_in[0];
    const float* norm_w   = (const float*)d_in[1];
    const float* in_projw = (const float*)d_in[2];
    const float* conv_w   = (const float*)d_in[3];
    const float* conv_b   = (const float*)d_in[4];
    const float* dt_bias  = (const float*)d_in[5];
    const float* A_log    = (const float*)d_in[6];
    const float* Dp       = (const float*)d_in[7];
    const float* gnw      = (const float*)d_in[8];
    const float* zero_w   = (const float*)d_in[9];
    const float* one_w    = (const float*)d_in[10];
    const float* fusion_w = (const float*)d_in[11];
    float* out = (float*)d_out;
    char*  ws  = (char*)d_ws;

    // workspace layout (bytes, 256B aligned), ~283 MB total
    constexpr size_t OFF_XN    = 0;
    constexpr size_t OFF_XNB   = al256(OFF_XN    + (size_t)ROWS * D_MODEL * 4);
    constexpr size_t OFF_WIN   = al256(OFF_XNB   + (size_t)ROWS * D_MODEL * 2);
    constexpr size_t OFF_WZER  = al256(OFF_WIN   + (size_t)D_IN_PROJ * D_MODEL * 2);
    constexpr size_t OFF_WONE  = al256(OFF_WZER  + (size_t)D_INNER * D_MODEL * 2);
    constexpr size_t OFF_WFUS  = al256(OFF_WONE  + (size_t)D_INNER * D_MODEL * 2);
    constexpr size_t OFF_ZX    = al256(OFF_WFUS  + (size_t)D_MODEL * DFUSE * 2);
    constexpr size_t OFF_XBC   = al256(OFF_ZX    + (size_t)ROWS * D_IN_PROJ * 4);
    constexpr size_t OFF_DT    = al256(OFF_XBC   + (size_t)ROWS * CONV_CH * 4);
    constexpr size_t OFF_DA    = al256(OFF_DT    + (size_t)ROWS * NHEADS * 4);
    constexpr size_t OFF_YS    = al256(OFF_DA    + (size_t)ROWS * NHEADS * 4);
    constexpr size_t OFF_ONE   = al256(OFF_YS    + (size_t)ROWS * D_INNER * 4);
    constexpr size_t OFF_FUSED = al256(OFF_ONE   + (size_t)ROWS * D_INNER * 4);

    float*          xn     = (float*)         (ws + OFF_XN);
    unsigned short* xnb    = (unsigned short*)(ws + OFF_XNB);
    unsigned short* w_in   = (unsigned short*)(ws + OFF_WIN);
    unsigned short* w_zero = (unsigned short*)(ws + OFF_WZER);
    unsigned short* w_one  = (unsigned short*)(ws + OFF_WONE);
    unsigned short* w_fuse = (unsigned short*)(ws + OFF_WFUS);
    float*          zx     = (float*)         (ws + OFF_ZX);
    float*          xbc    = (float*)         (ws + OFF_XBC);
    float*          dtb    = (float*)         (ws + OFF_DT);
    float*          dab    = (float*)         (ws + OFF_DA);
    float*          ys     = (float*)         (ws + OFF_YS);
    float*          oneo   = (float*)         (ws + OFF_ONE);
    unsigned short* fused  = (unsigned short*)(ws + OFF_FUSED);

    const dim3 blk(256);
    auto cvt = [&](const float* src, unsigned short* dst, int n) {
        cvt_bf16_kernel<<<dim3((n + 255) / 256), blk, 0, stream>>>(src, dst, n);
    };

    // 1) weight conversions to bf16 (weights then live in L2 across GEMMs)
    cvt(in_projw, w_in,   D_IN_PROJ * D_MODEL);
    cvt(zero_w,   w_zero, D_INNER * D_MODEL);
    cvt(one_w,    w_one,  D_INNER * D_MODEL);
    cvt(fusion_w, w_fuse, D_MODEL * DFUSE);

    // 2) input rmsnorm (f32 + bf16 copies)
    rmsnorm_in_kernel<<<dim3(ROWS), blk, 0, stream>>>(x, norm_w, xn, xnb);

    // 3) WMMA GEMMs fed by xnb   (grid = N/32, M/256)
    gemm_bf16_wmma<<<dim3(D_IN_PROJ / 32, ROWS / 256), blk, 0, stream>>>(
        xnb, w_in, D_MODEL, zx, nullptr, D_IN_PROJ, 0, nullptr);
    gemm_bf16_wmma<<<dim3(D_INNER / 32, ROWS / 256), blk, 0, stream>>>(
        xnb, w_zero, D_MODEL, nullptr, fused, DFUSE, D_INNER, nullptr);   // y_zero -> bf16 slice
    gemm_bf16_wmma<<<dim3(D_INNER / 32, ROWS / 256), blk, 0, stream>>>(
        xnb, w_one, D_MODEL, oneo, nullptr, D_INNER, 0, nullptr);

    // 4) conv + SiLU, dt/dA precompute
    {
        size_t n = (size_t)ROWS * CONV_CH;
        conv_silu_kernel<<<dim3((unsigned)((n + 255) / 256)), blk, 0, stream>>>(
            zx, conv_w, conv_b, xbc);
    }
    dtda_kernel<<<dim3((ROWS * NHEADS + 255) / 256), blk, 0, stream>>>(
        zx, dt_bias, A_log, dtb, dab);

    // 5) sequential SSM scan (64 independent (b,h) workgroups)
    ssm_scan_kernel<<<dim3(BATCH * NHEADS), blk, 0, stream>>>(xbc, dtb, dab, Dp, ys);

    // 6) gated rmsnorm -> fused[:, 0:2048)
    gate_norm_kernel<<<dim3(ROWS), blk, 0, stream>>>(ys, zx, gnw, fused);

    // 7) cumsum(one) -> fused[:, 4096:6144)
    cumsum_kernel<<<dim3((BATCH * D_INNER + 255) / 256), blk, 0, stream>>>(oneo, fused);

    // 8) fusion GEMM + residual -> out
    gemm_bf16_wmma<<<dim3(D_MODEL / 32, ROWS / 256), blk, 0, stream>>>(
        fused, w_fuse, DFUSE, out, nullptr, D_MODEL, 0, x);
}
